// EncoderEdgeGNN_25202868093637
// MI455X (gfx1250) — matrix operand order repaired and checked by reading, not compiled, not run on hardware
//
#include <hip/hip_runtime.h>

typedef __attribute__((ext_vector_type(16))) _Float16 v16h;
typedef __attribute__((ext_vector_type(8)))  _Float16 v8h;
typedef __attribute__((ext_vector_type(8)))  float    v8f;

static constexpr int NPG = 32, NGR = 64, NN = NPG * NGR;       // 2048 nodes
static constexpr int SD = 256, VD = 64, LATD = 128;
static constexpr int RBFD = 32, ED = 32, LAYERS = 5, HIDD = 256;
static constexpr int EIN = 2 * SD + RBFD + ED;                  // 576
static constexpr int EOUT = SD + 2 * VD;                        // 384
static constexpr int EPG = NPG * (NPG - 1);                     // 992 global edges / graph
static constexpr int TM  = 32;                                  // edge rows per block

__device__ __forceinline__ float silu_f(float x) { return x / (1.0f + __expf(-x)); }

__device__ __forceinline__ v8f wmma16(v16h a, v16h b, v8f c) {
  return __builtin_amdgcn_wmma_f32_16x16x32_f16(false, a, false, b, (short)0, c, false, false);
}

// ---------------------------------------------------------------- utility
__global__ void k_zero(float* __restrict__ p, int n) {
  int i = blockIdx.x * blockDim.x + threadIdx.x;
  if (i < n) p[i] = 0.0f;
}

// ------------------------------------------------ center positions per graph
__global__ void k_center(const float* __restrict__ pos, float* __restrict__ posc) {
  __shared__ float sm[3][NPG];
  int b = blockIdx.x, t = threadIdx.x;
  int n = b * NPG + t;
  float p0 = pos[n * 3 + 0], p1 = pos[n * 3 + 1], p2 = pos[n * 3 + 2];
  sm[0][t] = p0; sm[1][t] = p1; sm[2][t] = p2;
  __syncthreads();
  for (int off = NPG / 2; off > 0; off >>= 1) {
    if (t < off) { sm[0][t] += sm[0][t + off]; sm[1][t] += sm[1][t + off]; sm[2][t] += sm[2][t + off]; }
    __syncthreads();
  }
  float inv = 1.0f / (float)NPG;
  posc[n * 3 + 0] = p0 - sm[0][0] * inv;
  posc[n * 3 + 1] = p1 - sm[1][0] * inv;
  posc[n * 3 + 2] = p2 - sm[2][0] * inv;
}

// ------------------------------------------------ atom embedding s = x@W + b
__global__ void k_embed(const float* __restrict__ x, const float* __restrict__ aW,
                        const float* __restrict__ ab, float* __restrict__ s) {
  int n = blockIdx.x, c = threadIdx.x;
  float acc = ab[c];
  #pragma unroll
  for (int k = 0; k < 16; ++k) acc += x[n * 16 + k] * aW[k * SD + c];
  s[n * SD + c] = acc;
}

// ------------------------------------------------ bond embedding eA
__global__ void k_edge_embed(const float* __restrict__ ea, const float* __restrict__ bW,
                             const float* __restrict__ bb, int Eg, float* __restrict__ eA) {
  int idx = blockIdx.x * blockDim.x + threadIdx.x;
  if (idx >= Eg * ED) return;
  int e = idx / ED, c = idx % ED;
  float acc = bb[c];
  #pragma unroll
  for (int k = 0; k < 5; ++k) acc += ea[e * 5 + k] * bW[k * ED + c];
  eA[idx] = acc;
}

// ------------------------------------------------ geometry: rn + rbf per edge
__global__ void k_geom(const float* __restrict__ posc, const int* __restrict__ eg, int Eg,
                       float* __restrict__ rn, float* __restrict__ rbf) {
  int e = blockIdx.x * blockDim.x + threadIdx.x;
  if (e >= Eg) return;
  int src = eg[e], tgt = eg[Eg + e];
  float r0 = posc[tgt * 3 + 0] - posc[src * 3 + 0];
  float r1 = posc[tgt * 3 + 1] - posc[src * 3 + 1];
  float r2 = posc[tgt * 3 + 2] - posc[src * 3 + 2];
  float d2 = r0 * r0 + r1 * r1 + r2 * r2;
  float d = sqrtf(fmaxf(d2, 1e-6f));
  float inv = 1.0f / d;
  rn[e * 3 + 0] = r0 * inv; rn[e * 3 + 1] = r1 * inv; rn[e * 3 + 2] = r2 * inv;
  const float step = 7.5f / 31.0f;
  const float gamma = 1.0f / (2.0f * (7.5f / 32.0f) * (7.5f / 32.0f));
  #pragma unroll
  for (int i = 0; i < RBFD; ++i) {
    float dd = d - step * (float)i;
    rbf[e * RBFD + i] = __expf(-gamma * dd * dd);
  }
}

// ------------------------------------------------ local-edge aggregation (seg sum over tgt)
__global__ void k_local_agg(const float* __restrict__ eA, const int* __restrict__ el, int El,
                            float* __restrict__ locacc) {
  int idx = blockIdx.x * blockDim.x + threadIdx.x;
  if (idx >= El * ED) return;
  int e = idx / ED, c = idx % ED;
  int src = el[e], tgt = el[El + e];
  int b = src / NPG, i = src % NPG, j = tgt % NPG;
  int within = i * (NPG - 1) + (j - (j > i ? 1 : 0));   // all-pairs edge ordering
  int gidx = b * EPG + within;
  atomicAdd(&locacc[tgt * ED + c], eA[gidx * ED + c]);
}

// ------------------------------------------------ s += mean(locacc) @ loc_W
__global__ void k_add_loc(const float* __restrict__ locacc, const float* __restrict__ locW,
                          float* __restrict__ s) {
  int n = blockIdx.x, c = threadIdx.x;
  int i = n % NPG;
  float cnt = (float)(min(i, 4) + min(NPG - 1 - i, 4));
  float inv = 1.0f / fmaxf(cnt, 1.0f);
  float acc = 0.0f;
  #pragma unroll
  for (int k = 0; k < ED; ++k) acc += locacc[n * ED + k] * locW[k * SD + c];
  s[n * SD + c] += acc * inv;
}

// ------------------------------------------------ weight prep: f32 -> f16, transpose to [N][K]
__global__ void k_prep_w(const float* __restrict__ w1, const float* __restrict__ w2,
                         _Float16* __restrict__ w1t, _Float16* __restrict__ w2t) {
  int i = blockIdx.x * blockDim.x + threadIdx.x;
  const int N1 = LAYERS * EIN * HIDD;
  const int N2 = LAYERS * HIDD * EOUT;
  if (i < N1) {
    int l = i / (EIN * HIDD); int r = i % (EIN * HIDD);
    int k = r / HIDD; int n = r % HIDD;
    w1t[(size_t)l * HIDD * EIN + (size_t)n * EIN + k] = (_Float16)w1[i];
  } else if (i < N1 + N2) {
    int j = i - N1;
    int l = j / (HIDD * EOUT); int r = j % (HIDD * EOUT);
    int k = r / EOUT; int n = r % EOUT;
    w2t[(size_t)l * EOUT * HIDD + (size_t)n * HIDD + k] = (_Float16)w2[j];
  }
}

// ================================================ fused per-layer edge MLP (WMMA)
// 32-edge row tile per block; each B fragment feeds two row tiles (2x L2 weight reuse).
// Hs (32x592 f16 = 37.9KB) is dead after GEMM1 -> aliased with Mt (32x392 f32 = 50.2KB).
__global__ __launch_bounds__(256) void k_edge_layer(
    const float* __restrict__ s, const float* __restrict__ v,
    const float* __restrict__ rbf, const float* __restrict__ rn,
    const float* __restrict__ eA,
    const _Float16* __restrict__ W1t, const float* __restrict__ b1,
    const _Float16* __restrict__ W2t, const float* __restrict__ b2,
    const int* __restrict__ eg, int Eg,
    float* __restrict__ s_acc, float* __restrict__ v_acc)
{
  __shared__ __align__(16) char smem_u[TM * 392 * 4];   // Hs[TM][592] f16  UNION  Mt[TM][392] f32
  __shared__ __align__(16) _Float16 Hid[TM][264];       // 32 x 256 hidden (padded)
  __shared__ int   es[TM], et[TM];
  __shared__ float rns[TM][3];
  _Float16 (*Hs)[592] = reinterpret_cast<_Float16 (*)[592]>(smem_u);
  float    (*Mt)[392] = reinterpret_cast<float (*)[392]>(smem_u);

  const int tid = threadIdx.x;
  const int e0  = blockIdx.x * TM;

  if (tid < TM) {
    int e = e0 + tid;
    es[tid] = (e < Eg) ? eg[e] : 0;
    et[tid] = (e < Eg) ? eg[Eg + e] : 0;
  }
  if (tid < 3 * TM) {
    int r = tid / 3, d = tid % 3;
    int e = e0 + r;
    rns[r][d] = (e < Eg) ? rn[e * 3 + d] : 0.0f;
  }
  __syncthreads();

  // ---- gather + f16 conversion of H = [s[src] | s[tgt] | rbf | eA] ----
  for (int idx = tid; idx < TM * EIN; idx += 256) {
    int r = idx / EIN, c = idx % EIN;
    int e = e0 + r;
    float val = 0.0f;
    if (e < Eg) {
      if (c < SD)                 val = s[(size_t)es[r] * SD + c];
      else if (c < 2 * SD)        val = s[(size_t)et[r] * SD + (c - SD)];
      else if (c < 2 * SD + RBFD) val = rbf[(size_t)e * RBFD + (c - 2 * SD)];
      else                        val = eA[(size_t)e * ED + (c - 2 * SD - RBFD)];
    }
    Hs[r][c] = (_Float16)val;
  }
  __syncthreads();

  const int wave = tid >> 5;
  const int lane = tid & 31;
  const int m    = lane & 15;     // row-in-tile (A) / N-column-within-tile (B)
  const int hi   = lane >> 4;     // half-wave selector

  // ---- GEMM1: [32x576]f16 x [576x256]f16 -> SiLU -> Hid f16 ----
  {
    v8f c00 = {}, c01 = {}, c10 = {}, c11 = {};
    const int n0 = wave * 32 + m;                       // tile0 col, tile1 = +16
    for (int kc = 0; kc < EIN / 32; ++kc) {
      union { v16h vv; v8h h[2]; } ua0, ua1;
      int kb = kc * 32 + hi * 16;                       // A-layout: lanes 0-15 K=0-15, 16-31 K=16-31 (matches B)
      ua0.h[0] = *(const v8h*)(&Hs[m][kb]);
      ua0.h[1] = *(const v8h*)(&Hs[m][kb + 8]);
      ua1.h[0] = *(const v8h*)(&Hs[m + 16][kb]);
      ua1.h[1] = *(const v8h*)(&Hs[m + 16][kb + 8]);
      int kB = kc * 32 + hi * 16;                       // B-layout: lanes 0-15 K=0-15, 16-31 K=16-31 (contiguous, same as A)
      const _Float16* brow0 = W1t + (size_t)n0 * EIN + kB;
      const _Float16* brow1 = W1t + (size_t)(n0 + 16) * EIN + kB;
      if (kc + 1 < EIN / 32) {                          // global_prefetch_b8 on next K chunk
        __builtin_prefetch((const void*)(brow0 + 32), 0, 3);
        __builtin_prefetch((const void*)(brow1 + 32), 0, 3);
      }
      v16h bf0 = *(const v16h*)brow0;
      v16h bf1 = *(const v16h*)brow1;
      c00 = wmma16(ua0.vv, bf0, c00);
      c01 = wmma16(ua0.vv, bf1, c01);
      c10 = wmma16(ua1.vv, bf0, c10);
      c11 = wmma16(ua1.vv, bf1, c11);
    }
    const int colb = wave * 32 + m;
    #pragma unroll
    for (int r = 0; r < 8; ++r) {
      int mr = r + hi * 8;                              // ISA C/D layout
      Hid[mr][colb]           = (_Float16)silu_f(c00[r] + b1[colb]);
      Hid[mr][colb + 16]      = (_Float16)silu_f(c01[r] + b1[colb + 16]);
      Hid[mr + 16][colb]      = (_Float16)silu_f(c10[r] + b1[colb]);
      Hid[mr + 16][colb + 16] = (_Float16)silu_f(c11[r] + b1[colb + 16]);
    }
  }
  __syncthreads();   // all waves done with Hs; Mt (aliased) may now be written

  // ---- GEMM2: [32x256]f16 x [256x384]f16 -> Mt f32 ----
  {
    v8f d00 = {}, d01 = {}, d02 = {}, d10 = {}, d11 = {}, d12 = {};
    const int nb = wave * 48;
    for (int kc = 0; kc < HIDD / 32; ++kc) {
      union { v16h vv; v8h h[2]; } ua0, ua1;
      int kb = kc * 32 + hi * 16;
      ua0.h[0] = *(const v8h*)(&Hid[m][kb]);
      ua0.h[1] = *(const v8h*)(&Hid[m][kb + 8]);
      ua1.h[0] = *(const v8h*)(&Hid[m + 16][kb]);
      ua1.h[1] = *(const v8h*)(&Hid[m + 16][kb + 8]);
      int kB = kc * 32 + hi * 16;
      v16h bf0 = *(const v16h*)(W2t + (size_t)(nb +  0 + m) * HIDD + kB);
      v16h bf1 = *(const v16h*)(W2t + (size_t)(nb + 16 + m) * HIDD + kB);
      v16h bf2 = *(const v16h*)(W2t + (size_t)(nb + 32 + m) * HIDD + kB);
      d00 = wmma16(ua0.vv, bf0, d00);
      d01 = wmma16(ua0.vv, bf1, d01);
      d02 = wmma16(ua0.vv, bf2, d02);
      d10 = wmma16(ua1.vv, bf0, d10);
      d11 = wmma16(ua1.vv, bf1, d11);
      d12 = wmma16(ua1.vv, bf2, d12);
    }
    #pragma unroll
    for (int r = 0; r < 8; ++r) {
      int mr = r + hi * 8;
      Mt[mr][nb +  0 + m]      = d00[r] + b2[nb +  0 + m];
      Mt[mr][nb + 16 + m]      = d01[r] + b2[nb + 16 + m];
      Mt[mr][nb + 32 + m]      = d02[r] + b2[nb + 32 + m];
      Mt[mr + 16][nb +  0 + m] = d10[r] + b2[nb +  0 + m];
      Mt[mr + 16][nb + 16 + m] = d11[r] + b2[nb + 16 + m];
      Mt[mr + 16][nb + 32 + m] = d12[r] + b2[nb + 32 + m];
    }
  }
  __syncthreads();

  // ---- scatter ms (seg sum over tgt, count==31 handled at finalize) ----
  for (int idx = tid; idx < TM * SD; idx += 256) {
    int r = idx >> 8, c = idx & 255;
    if (e0 + r < Eg) atomicAdd(&s_acc[(size_t)et[r] * SD + c], Mt[r][c]);
  }
  // ---- scatter vmsg = rn*mv1 + v[src]*mv2 ----
  for (int idx = tid; idx < TM * 3 * VD; idx += 256) {
    int r = idx / (3 * VD);
    int rem = idx % (3 * VD);
    int d = rem / VD, j = rem % VD;
    if (e0 + r < Eg) {
      float mv1 = Mt[r][SD + j];
      float mv2 = Mt[r][SD + VD + j];
      float val = rns[r][d] * mv1 + v[(size_t)es[r] * (3 * VD) + d * VD + j] * mv2;
      atomicAdd(&v_acc[(size_t)et[r] * (3 * VD) + d * VD + j], val);
    }
  }
}

// ------------------------------------------------ finalize layer: s,v update + LayerNorm(s)
__global__ __launch_bounds__(256) void k_node_update(
    float* __restrict__ s, float* __restrict__ v,
    const float* __restrict__ s_acc, const float* __restrict__ v_acc,
    const float* __restrict__ lng, const float* __restrict__ lnb)
{
  __shared__ float red[256];
  __shared__ float mu_s, var_s;
  int n = blockIdx.x, t = threadIdx.x;
  const float invdeg = 1.0f / (float)(NPG - 1);   // global-edge in-degree is exactly 31
  float sv = s[(size_t)n * SD + t] + s_acc[(size_t)n * SD + t] * invdeg;
  if (t < 3 * VD)
    v[(size_t)n * 3 * VD + t] += v_acc[(size_t)n * 3 * VD + t] * invdeg;
  red[t] = sv; __syncthreads();
  for (int off = 128; off > 0; off >>= 1) { if (t < off) red[t] += red[t + off]; __syncthreads(); }
  if (t == 0) mu_s = red[0] * (1.0f / SD);
  __syncthreads();
  float df = sv - mu_s;
  red[t] = df * df; __syncthreads();
  for (int off = 128; off > 0; off >>= 1) { if (t < off) red[t] += red[t + off]; __syncthreads(); }
  if (t == 0) var_s = red[0] * (1.0f / SD);
  __syncthreads();
  s[(size_t)n * SD + t] = df * rsqrtf(var_s + 1e-5f) * lng[t] + lnb[t];
}

// ------------------------------------------------ readout MLPs per node
__global__ __launch_bounds__(128) void k_readout(
    const float* __restrict__ s,
    const float* __restrict__ latW, const float* __restrict__ latb,
    const float* __restrict__ nW1, const float* __restrict__ nb1,
    const float* __restrict__ nW2, const float* __restrict__ nb2,
    const float* __restrict__ gW1, const float* __restrict__ gb1,
    const float* __restrict__ gW2, const float* __restrict__ gb2,
    float* __restrict__ gbuf, float* __restrict__ nodebuf)
{
  __shared__ float lat[LATD], gh[LATD], nh[LATD], red[LATD];
  int n = blockIdx.x, t = threadIdx.x;
  float acc = latb[t];
  for (int k = 0; k < SD; ++k) acc += s[(size_t)n * SD + k] * latW[k * LATD + t];
  lat[t] = acc; __syncthreads();
  float ag = gb1[t], an = nb1[t];
  for (int k = 0; k < LATD; ++k) { float lv = lat[k]; ag += lv * gW1[k * LATD + t]; an += lv * nW1[k * LATD + t]; }
  gh[t] = silu_f(ag); nh[t] = silu_f(an); __syncthreads();
  float no = nb2[t];
  for (int k = 0; k < LATD; ++k) no += nh[k] * nW2[k * LATD + t];
  nodebuf[(size_t)n * LATD + t] = no;
  red[t] = gh[t] * gW2[t]; __syncthreads();
  for (int off = 64; off > 0; off >>= 1) { if (t < off) red[t] += red[t + off]; __syncthreads(); }
  if (t == 0) gbuf[n] = red[0] + gb2[0];
}

// ------------------------------------------------ per-graph softmax pooling
__global__ __launch_bounds__(128) void k_pool(const float* __restrict__ gbuf,
                                              const float* __restrict__ nodebuf,
                                              float* __restrict__ out)
{
  __shared__ float w[NPG];
  int b = blockIdx.x, t = threadIdx.x;
  if (t == 0) {
    float mx = -1e30f;
    for (int i = 0; i < NPG; ++i) mx = fmaxf(mx, gbuf[b * NPG + i]);
    float den = 0.0f;
    for (int i = 0; i < NPG; ++i) { float e = __expf(gbuf[b * NPG + i] - mx); w[i] = e; den += e; }
    float inv = 1.0f / den;
    for (int i = 0; i < NPG; ++i) w[i] *= inv;
  }
  __syncthreads();
  float acc = 0.0f;
  for (int i = 0; i < NPG; ++i) acc += w[i] * nodebuf[(size_t)(b * NPG + i) * LATD + t];
  out[b * LATD + t] = acc;
}

// ================================================================ launch
extern "C" void kernel_launch(void* const* d_in, const int* in_sizes, int n_in,
                              void* d_out, int out_size, void* d_ws, size_t ws_size,
                              hipStream_t stream) {
  const float* x       = (const float*)d_in[0];
  const float* pos     = (const float*)d_in[1];
  const float* eattr   = (const float*)d_in[2];
  const float* atom_W  = (const float*)d_in[3];
  const float* atom_b  = (const float*)d_in[4];
  const float* bond_W  = (const float*)d_in[5];
  const float* bond_b  = (const float*)d_in[6];
  const float* loc_W   = (const float*)d_in[7];
  const float* l_W1    = (const float*)d_in[8];
  const float* l_b1    = (const float*)d_in[9];
  const float* l_W2    = (const float*)d_in[10];
  const float* l_b2    = (const float*)d_in[11];
  const float* ln_g    = (const float*)d_in[12];
  const float* ln_b    = (const float*)d_in[13];
  const float* lat_W   = (const float*)d_in[14];
  const float* lat_b   = (const float*)d_in[15];
  const float* node_W1 = (const float*)d_in[16];
  const float* node_b1 = (const float*)d_in[17];
  const float* node_W2 = (const float*)d_in[18];
  const float* node_b2 = (const float*)d_in[19];
  const float* gate_W1 = (const float*)d_in[20];
  const float* gate_b1 = (const float*)d_in[21];
  const float* gate_W2 = (const float*)d_in[22];
  const float* gate_b2 = (const float*)d_in[23];
  const int*   el      = (const int*)d_in[24];
  const int*   eg      = (const int*)d_in[25];
  (void)n_in; (void)out_size; (void)ws_size;

  const int El = in_sizes[24] / 2;
  const int Eg = in_sizes[25] / 2;

  // carve workspace (256B-aligned regions)
  char* p = (char*)d_ws;
  auto carve = [&](size_t bytes) -> char* {
    char* r = p; p += (bytes + 255) & ~(size_t)255; return r;
  };
  float*    posc    = (float*)carve(sizeof(float) * NN * 3);
  float*    s       = (float*)carve(sizeof(float) * NN * SD);
  float*    v       = (float*)carve(sizeof(float) * NN * 3 * VD);
  float*    s_acc   = (float*)carve(sizeof(float) * NN * SD);
  float*    v_acc   = (float*)carve(sizeof(float) * NN * 3 * VD);
  float*    eAb     = (float*)carve(sizeof(float) * (size_t)Eg * ED);
  float*    rbfb    = (float*)carve(sizeof(float) * (size_t)Eg * RBFD);
  float*    rnb     = (float*)carve(sizeof(float) * (size_t)Eg * 3);
  float*    locacc  = (float*)carve(sizeof(float) * NN * ED);
  float*    gbuf    = (float*)carve(sizeof(float) * NN);
  float*    nodebuf = (float*)carve(sizeof(float) * NN * LATD);
  _Float16* W1t     = (_Float16*)carve(sizeof(_Float16) * LAYERS * HIDD * EIN);
  _Float16* W2t     = (_Float16*)carve(sizeof(_Float16) * LAYERS * EOUT * HIDD);

  k_center<<<NGR, NPG, 0, stream>>>(pos, posc);
  k_zero<<<(NN * 3 * VD + 255) / 256, 256, 0, stream>>>(v, NN * 3 * VD);
  k_zero<<<(NN * ED + 255) / 256, 256, 0, stream>>>(locacc, NN * ED);
  k_embed<<<NN, SD, 0, stream>>>(x, atom_W, atom_b, s);
  k_edge_embed<<<(Eg * ED + 255) / 256, 256, 0, stream>>>(eattr, bond_W, bond_b, Eg, eAb);
  k_geom<<<(Eg + 255) / 256, 256, 0, stream>>>(posc, eg, Eg, rnb, rbfb);
  k_local_agg<<<(El * ED + 255) / 256, 256, 0, stream>>>(eAb, el, El, locacc);
  k_add_loc<<<NN, SD, 0, stream>>>(locacc, loc_W, s);
  {
    int total = LAYERS * EIN * HIDD + LAYERS * HIDD * EOUT;
    k_prep_w<<<(total + 255) / 256, 256, 0, stream>>>(l_W1, l_W2, W1t, W2t);
  }

  for (int l = 0; l < LAYERS; ++l) {
    k_zero<<<(NN * SD + 255) / 256, 256, 0, stream>>>(s_acc, NN * SD);
    k_zero<<<(NN * 3 * VD + 255) / 256, 256, 0, stream>>>(v_acc, NN * 3 * VD);
    k_edge_layer<<<(Eg + TM - 1) / TM, 256, 0, stream>>>(
        s, v, rbfb, rnb, eAb,
        W1t + (size_t)l * HIDD * EIN, l_b1 + l * HIDD,
        W2t + (size_t)l * EOUT * HIDD, l_b2 + l * EOUT,
        eg, Eg, s_acc, v_acc);
    k_node_update<<<NN, SD, 0, stream>>>(s, v, s_acc, v_acc, ln_g + l * SD, ln_b + l * SD);
  }

  k_readout<<<NN, LATD, 0, stream>>>(s, lat_W, lat_b, node_W1, node_b1, node_W2, node_b2,
                                     gate_W1, gate_b1, gate_W2, gate_b2, gbuf, nodebuf);
  k_pool<<<NGR, LATD, 0, stream>>>(gbuf, nodebuf, (float*)d_out);
}
